// GATLayer_10617159156297
// MI455X (gfx1250) — matrix operand order repaired and checked by reading, not compiled, not run on hardware
//
#include <hip/hip_runtime.h>

// Problem constants (match reference)
#define BB 8
#define NN 1024
#define DD 128
#define HH 8
#define DH 16
#define FF 512
#define CLIPV 10.0f

typedef __attribute__((ext_vector_type(16))) _Float16 v16h;
typedef __attribute__((ext_vector_type(8)))  float    v8f;

static __device__ __forceinline__ void wait_dscnt0() {
  asm volatile("s_wait_dscnt 0" ::: "memory");
}
static __device__ __forceinline__ void wait_asynccnt(void) {
  // previous tile's 4 async ops done when <=4 outstanding (in-order completion)
  asm volatile("s_wait_asynccnt 4" ::: "memory");
}
// DMA 32 bytes (one key row: 16 f16) global -> LDS, as 2 x b128.
// IOFFSET applies to both LDS and global address (ISA 08 §4.4).
static __device__ __forceinline__ void async_row32(unsigned lds_addr, const _Float16* g) {
  asm volatile("global_load_async_to_lds_b128 %0, %1, off" :: "v"(lds_addr), "v"(g) : "memory");
  asm volatile("global_load_async_to_lds_b128 %0, %1, off offset:16" :: "v"(lds_addr), "v"(g) : "memory");
}

static __device__ __forceinline__ v8f wmma32(v16h a, v16h b, v8f c) {
  // D = A(16x32 f16) * B(32x16 f16) + C(16x16 f32)
  return __builtin_amdgcn_wmma_f32_16x16x32_f16(
      /*neg_a=*/false, a, /*neg_b=*/false, b,
      /*c_mod=*/(short)0, c, /*reuse_a=*/false, /*reuse_b=*/false);
}

// ---- fragment loaders (contiguous k = 16*(lane/16) + s, wave32) -------------------------

// A-matrix 16x32 f16, row-major source with leading dim lda.
static __device__ __forceinline__ v16h load_a16(const _Float16* __restrict__ A, int lda) {
  int lane = threadIdx.x & 31;
  int m = lane & 15, h8 = (lane >> 4) * 8;
  const _Float16* p = A + m * lda;
  v16h a;
#pragma unroll
  for (int j = 0; j < 8; ++j) {
    int kk = (h8 << 1) + 2 * j;
    a[2 * j]     = p[kk];
    a[2 * j + 1] = p[kk + 1];
  }
  return a;
}

// A-frag, only K < kmax valid (rest zero) — used for dh=16 padded to K=32
static __device__ __forceinline__ v16h load_a16_pad(const _Float16* __restrict__ A, int lda, int kmax) {
  int lane = threadIdx.x & 31;
  int m = lane & 15, h8 = (lane >> 4) * 8;
  const _Float16* p = A + m * lda;
  v16h a;
#pragma unroll
  for (int j = 0; j < 8; ++j) {
    int kk = (h8 << 1) + 2 * j;
    a[2 * j]     = (kk     < kmax) ? p[kk]     : (_Float16)0.0f;
    a[2 * j + 1] = (kk + 1 < kmax) ? p[kk + 1] : (_Float16)0.0f;
  }
  return a;
}

// B-matrix 32x16 f16, B[k][n] = src[k*ldb + n] (row-major KxN).
static __device__ __forceinline__ v16h load_b16(const _Float16* __restrict__ Bm, int ldb) {
  int lane = threadIdx.x & 31;
  int n = lane & 15, h16 = (lane >> 4) * 16;
  v16h b;
#pragma unroll
  for (int j = 0; j < 8; ++j) {
    int kk = h16 + 2 * j;
    b[2 * j]     = Bm[kk * ldb + n];
    b[2 * j + 1] = Bm[(kk + 1) * ldb + n];
  }
  return b;
}

// B-frag from f32 row-major source with on-the-fly f32->f16 convert (weights)
static __device__ __forceinline__ v16h load_b32cvt(const float* __restrict__ Bm, int ldb) {
  int lane = threadIdx.x & 31;
  int n = lane & 15, h16 = (lane >> 4) * 16;
  v16h b;
#pragma unroll
  for (int j = 0; j < 8; ++j) {
    int kk = h16 + 2 * j;
    b[2 * j]     = (_Float16)Bm[kk * ldb + n];
    b[2 * j + 1] = (_Float16)Bm[(kk + 1) * ldb + n];
  }
  return b;
}

// B-frag where logical B[k][n] = src[n*ldr + k] (K^T), only K < kmax valid.
static __device__ __forceinline__ v16h load_bT16_pad(const _Float16* __restrict__ Bm, int ldr, int kmax) {
  int lane = threadIdx.x & 31;
  int n = lane & 15, h16 = (lane >> 4) * 16;
  v16h b;
#pragma unroll
  for (int j = 0; j < 8; ++j) {
    int kk = h16 + 2 * j;
    b[2 * j]     = (kk     < kmax) ? Bm[n * ldr + kk]     : (_Float16)0.0f;
    b[2 * j + 1] = (kk + 1 < kmax) ? Bm[n * ldr + kk + 1] : (_Float16)0.0f;
  }
  return b;
}

// ---- kernels ---------------------------------------------------------------

__global__ __launch_bounds__(256) void k_cvt(const float* __restrict__ x,
                                             _Float16* __restrict__ xh, int n) {
  int i = blockIdx.x * 256 + threadIdx.x;
  if (i < n) xh[i] = (_Float16)x[i];
}

// Q/K/V projection: [8192 x 128] @ [128 x 128] three times.
__global__ __launch_bounds__(32) void k_proj(const _Float16* __restrict__ xh,
                                             const float* __restrict__ Wq,
                                             const float* __restrict__ Wk,
                                             const float* __restrict__ Wv,
                                             _Float16* __restrict__ q,
                                             _Float16* __restrict__ k,
                                             _Float16* __restrict__ v) {
  int nt = blockIdx.x, which = blockIdx.y, slab = blockIdx.z;
  const float* W = (which == 0) ? Wq : (which == 1) ? Wk : Wv;
  _Float16* out = (which == 0) ? q : (which == 1) ? k : v;
  int lane = threadIdx.x, n = lane & 15, half = lane >> 4;
  int col0 = nt * 16;

  v16h bf[4];
#pragma unroll
  for (int kb = 0; kb < 4; ++kb) bf[kb] = load_b32cvt(W + (kb * 32) * DD + col0, DD);

  for (int mt = 0; mt < 16; ++mt) {
    int row0 = (slab * 16 + mt) * 16;
    v8f acc = {};
#pragma unroll
    for (int kb = 0; kb < 4; ++kb) {
      v16h a = load_a16(xh + (size_t)row0 * DD + kb * 32, DD);
      acc = wmma32(a, bf[kb], acc);
    }
    _Float16* op = out + (size_t)(row0 + half * 8) * DD + col0 + n;
#pragma unroll
    for (int r = 0; r < 8; ++r) op[r * DD] = (_Float16)acc[r];
  }
}

// Flash attention per (b, 16-row tile), 8 waves = 8 heads per block.
// K/V tiles are DMA'd global->LDS with async b128 ops, double buffered so the
// DMA of tile t+1 overlaps tanh/exp/WMMA on tile t.
__global__ __launch_bounds__(256) void k_attn(const _Float16* __restrict__ q,
                                              const _Float16* __restrict__ k,
                                              const _Float16* __restrict__ v,
                                              const float* __restrict__ e,
                                              _Float16* __restrict__ ao) {
  __shared__ _Float16 kbuf[HH][2][32 * DH];  // 32 keys x 16 feat, double buffered (1KB each)
  __shared__ _Float16 vbuf[HH][2][32 * DH];
  __shared__ _Float16 pbuf[HH][16 * 32];     // P tile C-layout -> A-layout transpose
  int mtile = blockIdx.x;                    // 0..63
  int b = blockIdx.y;                        // 0..7
  int h = threadIdx.x >> 5;                  // wave id = head
  int lane = threadIdx.x & 31, n = lane & 15, half = lane >> 4;
  int row0 = mtile * 16;
  _Float16* pb = &pbuf[h][0];

  // per-lane LDS byte addresses: lane owns key row `lane` (32 bytes) of each buffer
  unsigned kd[2] = { (unsigned)(size_t)&kbuf[h][0][lane * DH],
                     (unsigned)(size_t)&kbuf[h][1][lane * DH] };
  unsigned vd[2] = { (unsigned)(size_t)&vbuf[h][0][lane * DH],
                     (unsigned)(size_t)&vbuf[h][1][lane * DH] };
  const _Float16* kgbase = k + ((size_t)b * NN) * DD + h * DH;
  const _Float16* vgbase = v + ((size_t)b * NN) * DD + h * DH;

  // Q fragment: 16 rows x dh=16, K-padded to 32
  v16h qa = load_a16_pad(q + ((size_t)b * NN + row0) * DD + h * DH, DD, DH);

  v8f acc = {};
  float mrow[8], lrow[8];
#pragma unroll
  for (int r = 0; r < 8; ++r) { mrow[r] = -1e30f; lrow[r] = 0.0f; }

  const float* erow = e + ((size_t)b * NN + row0 + half * 8) * NN;

  // prologue: DMA tile 0 into buffer 0
  async_row32(kd[0], kgbase + (size_t)lane * DD);
  async_row32(vd[0], vgbase + (size_t)lane * DD);

  for (int t = 0; t < NN / 32; ++t) {
    int cur = t & 1;
    int key0 = t * 32;
    // DMA next tile (wraps harmlessly at the end) into the other buffer
    int tn = (t + 1) & (NN / 32 - 1);
    async_row32(kd[cur ^ 1], kgbase + (size_t)(tn * 32 + lane) * DD);
    async_row32(vd[cur ^ 1], vgbase + (size_t)(tn * 32 + lane) * DD);
    wait_asynccnt();                         // tile t resident (in-order, <=4 outstanding)

    const _Float16* kcur = &kbuf[h][cur][0];
    const _Float16* vcur = &vbuf[h][cur][0];

    // K^T fragments: B[kk][n] = kcur[n*DH + kk], 2 groups of 16 keys
    v16h kf0 = load_bT16_pad(kcur, DH, DH);
    v16h kf1 = load_bT16_pad(kcur + 16 * DH, DH, DH);
    v8f z = {};
    v8f s0 = wmma32(qa, kf0, z);
    v8f s1 = wmma32(qa, kf1, z);

#pragma unroll
    for (int r = 0; r < 8; ++r) {
      float e0 = erow[(size_t)r * NN + key0 + n];
      float e1 = erow[(size_t)r * NN + key0 + 16 + n];
      float t0 = CLIPV * tanhf(s0[r] * 0.25f + e0);   // 1/sqrt(16)=0.25
      float t1 = CLIPV * tanhf(s1[r] * 0.25f + e1);
      float tm = fmaxf(t0, t1);
      tm = fmaxf(tm, __shfl_xor(tm, 1, 32));
      tm = fmaxf(tm, __shfl_xor(tm, 2, 32));
      tm = fmaxf(tm, __shfl_xor(tm, 4, 32));
      tm = fmaxf(tm, __shfl_xor(tm, 8, 32));
      float mnew = fmaxf(mrow[r], tm);
      float sc = __expf(mrow[r] - mnew);
      float p0 = __expf(t0 - mnew);
      float p1 = __expf(t1 - mnew);
      float ps = p0 + p1;
      ps += __shfl_xor(ps, 1, 32);
      ps += __shfl_xor(ps, 2, 32);
      ps += __shfl_xor(ps, 4, 32);
      ps += __shfl_xor(ps, 8, 32);
      lrow[r] = lrow[r] * sc + ps;
      mrow[r] = mnew;
      acc[r] *= sc;
      int rl = r + half * 8;
      pb[rl * 32 + n]      = (_Float16)p0;   // ds_store_b16
      pb[rl * 32 + 16 + n] = (_Float16)p1;
    }
    wait_dscnt0();                           // P stores visible before re-read

    v16h pf = load_a16(pb, 32);              // P as A-frag (LDS)
    v16h vf = load_b16(vcur, DH);            // V as B-frag from LDS, K=32 keys
    acc = wmma32(pf, vf, acc);
    wait_dscnt0();                           // LDS reads done before next DMA overwrites
  }

  _Float16* op = ao + ((size_t)b * NN + row0 + half * 8) * DD + h * DH + n;
#pragma unroll
  for (int r = 0; r < 8; ++r) op[r * DD] = (_Float16)(acc[r] / lrow[r]);
}

// x1 = x + alpha1 * (attnout @ Wo) ; emits f32 x1 and f16 x1h
__global__ __launch_bounds__(32) void k_outproj(const _Float16* __restrict__ ao,
                                                const float* __restrict__ Wo,
                                                const float* __restrict__ x,
                                                const float* __restrict__ a1p,
                                                float* __restrict__ x1,
                                                _Float16* __restrict__ x1h) {
  int nt = blockIdx.x, slab = blockIdx.y;
  int lane = threadIdx.x, n = lane & 15, half = lane >> 4;
  int col0 = nt * 16;
  float alpha = a1p[0];

  v16h bf[4];
#pragma unroll
  for (int kb = 0; kb < 4; ++kb) bf[kb] = load_b32cvt(Wo + (kb * 32) * DD + col0, DD);

  for (int mt = 0; mt < 16; ++mt) {
    int row0 = (slab * 16 + mt) * 16;
    v8f acc = {};
#pragma unroll
    for (int kb = 0; kb < 4; ++kb) {
      v16h a = load_a16(ao + (size_t)row0 * DD + kb * 32, DD);
      acc = wmma32(a, bf[kb], acc);
    }
    size_t base = (size_t)(row0 + half * 8) * DD + col0 + n;
#pragma unroll
    for (int r = 0; r < 8; ++r) {
      float val = x[base + (size_t)r * DD] + alpha * acc[r];
      x1[base + (size_t)r * DD] = val;
      x1h[base + (size_t)r * DD] = (_Float16)val;
    }
  }
}

// hidden = relu(x1h @ W1 + b1)  [8192 x 512] f16
__global__ __launch_bounds__(32) void k_ffn1(const _Float16* __restrict__ x1h,
                                             const float* __restrict__ W1,
                                             const float* __restrict__ b1,
                                             _Float16* __restrict__ hid) {
  int nt = blockIdx.x, slab = blockIdx.y;
  int lane = threadIdx.x, n = lane & 15, half = lane >> 4;
  int col0 = nt * 16;
  float bias = b1[col0 + n];

  v16h bf[4];
#pragma unroll
  for (int kb = 0; kb < 4; ++kb) bf[kb] = load_b32cvt(W1 + (kb * 32) * FF + col0, FF);

  for (int mt = 0; mt < 16; ++mt) {
    int row0 = (slab * 16 + mt) * 16;
    v8f acc = {};
#pragma unroll
    for (int kb = 0; kb < 4; ++kb) {
      v16h a = load_a16(x1h + (size_t)row0 * DD + kb * 32, DD);
      acc = wmma32(a, bf[kb], acc);
    }
    _Float16* op = hid + (size_t)(row0 + half * 8) * FF + col0 + n;
#pragma unroll
    for (int r = 0; r < 8; ++r) op[r * FF] = (_Float16)fmaxf(acc[r] + bias, 0.0f);
  }
}

// out = x1 + alpha2 * (hidden @ W2 + b2)   K=512: all 16 B-frags cached in VGPRs
__global__ __launch_bounds__(32) void k_ffn2(const _Float16* __restrict__ hid,
                                             const float* __restrict__ W2,
                                             const float* __restrict__ b2,
                                             const float* __restrict__ x1,
                                             const float* __restrict__ a2p,
                                             float* __restrict__ out) {
  int nt = blockIdx.x, slab = blockIdx.y;
  int lane = threadIdx.x, n = lane & 15, half = lane >> 4;
  int col0 = nt * 16;
  float alpha = a2p[0];
  float bias = b2[col0 + n];

  v16h bf[16];
#pragma unroll
  for (int kb = 0; kb < 16; ++kb) bf[kb] = load_b32cvt(W2 + (kb * 32) * DD + col0, DD);

  for (int mt = 0; mt < 16; ++mt) {
    int row0 = (slab * 16 + mt) * 16;
    v8f acc = {};
#pragma unroll
    for (int kb = 0; kb < 16; ++kb) {
      v16h a = load_a16(hid + (size_t)row0 * FF + kb * 32, FF);
      acc = wmma32(a, bf[kb], acc);
    }
    size_t base = (size_t)(row0 + half * 8) * DD + col0 + n;
#pragma unroll
    for (int r = 0; r < 8; ++r)
      out[base + (size_t)r * DD] = x1[base + (size_t)r * DD] + alpha * (acc[r] + bias);
  }
}

// ---- launch ---------------------------------------------------------------

extern "C" void kernel_launch(void* const* d_in, const int* in_sizes, int n_in,
                              void* d_out, int out_size, void* d_ws, size_t ws_size,
                              hipStream_t stream) {
  (void)in_sizes; (void)n_in; (void)out_size; (void)ws_size;
  const float* x  = (const float*)d_in[0];
  const float* e  = (const float*)d_in[1];
  const float* Wq = (const float*)d_in[2];
  const float* Wk = (const float*)d_in[3];
  const float* Wv = (const float*)d_in[4];
  const float* Wo = (const float*)d_in[5];
  const float* W1 = (const float*)d_in[6];
  const float* b1 = (const float*)d_in[7];
  const float* W2 = (const float*)d_in[8];
  const float* b2 = (const float*)d_in[9];
  const float* a1 = (const float*)d_in[10];
  const float* a2 = (const float*)d_in[11];
  float* out = (float*)d_out;

  const size_t MB = 1u << 20;
  char* ws = (char*)d_ws;
  _Float16* xh  = (_Float16*)(ws);             // 2 MB
  _Float16* q   = (_Float16*)(ws + 2 * MB);    // 2 MB
  _Float16* k   = (_Float16*)(ws + 4 * MB);    // 2 MB
  _Float16* v   = (_Float16*)(ws + 6 * MB);    // 2 MB
  _Float16* ao  = (_Float16*)(ws + 8 * MB);    // 2 MB
  float*    x1  = (float*)   (ws + 10 * MB);   // 4 MB
  _Float16* x1h = (_Float16*)(ws + 14 * MB);   // 2 MB
  _Float16* hid = (_Float16*)(ws + 16 * MB);   // 8 MB

  const int nrow = BB * NN;                    // 8192 rows

  k_cvt<<<(nrow * DD) / 256, 256, 0, stream>>>(x, xh, nrow * DD);
  k_proj<<<dim3(DD / 16, 3, nrow / 256), 32, 0, stream>>>(xh, Wq, Wk, Wv, q, k, v);
  k_attn<<<dim3(NN / 16, BB), 256, 0, stream>>>(q, k, v, e, ao);
  k_outproj<<<dim3(DD / 16, nrow / 256), 32, 0, stream>>>(ao, Wo, x, a1, x1, x1h);
  k_ffn1<<<dim3(FF / 16, nrow / 256), 32, 0, stream>>>(x1h, W1, b1, hid);
  k_ffn2<<<dim3(DD / 16, nrow / 256), 32, 0, stream>>>(hid, W2, b2, x1, a2, out);
}
